// PointNetPP_40484361732388
// MI455X (gfx1250) — compile-verified, matrix operand order checked by reading, not run on hardware
//
#include <hip/hip_runtime.h>

typedef __attribute__((ext_vector_type(16))) _Float16 v16h;
typedef __attribute__((ext_vector_type(8)))  _Float16 v8h;
typedef __attribute__((ext_vector_type(8)))  float    v8f;

#define BATCH 8
#define NPTS  4096
#define CIN   6
#define NCLS  13
#define M1    820
#define M2    205
#define KNB   16
#define BN_SCALE_F 0.9999950000374997f   // 1/sqrt(1+1e-5)

// ---------------------------------------------------------------------------
// Farthest point sampling: one block per batch cloud, mind[] lives in LDS.
// ---------------------------------------------------------------------------
template<int NT>
__global__ __launch_bounds__(NT)
void fps_kernel(const float* __restrict__ pos, int Nn, int M,
                int* __restrict__ idx, float* __restrict__ centers)
{
  __shared__ float mind[4096];
  __shared__ float redv[NT / 32];
  __shared__ int   redi[NT / 32];
  __shared__ float cpt[3];
  const int b   = blockIdx.x;
  const float* P = pos + (size_t)b * Nn * 3;
  const int tid = threadIdx.x;
  for (int i = tid; i < Nn; i += NT) mind[i] = 1e10f;
  if (tid == 0) {
    idx[b * M] = 0;
    float px = P[0], py = P[1], pz = P[2];
    centers[(size_t)(b * M) * 3 + 0] = px;
    centers[(size_t)(b * M) * 3 + 1] = py;
    centers[(size_t)(b * M) * 3 + 2] = pz;
    cpt[0] = px; cpt[1] = py; cpt[2] = pz;
  }
  __syncthreads();
  for (int it = 1; it < M; ++it) {
    const float cx = cpt[0], cy = cpt[1], cz = cpt[2];
    float bv = -1.f; int bi = 0x7FFFFFFF;
    for (int i = tid; i < Nn; i += NT) {
      float dx = P[i*3+0]-cx, dy = P[i*3+1]-cy, dz = P[i*3+2]-cz;
      float d  = dx*dx + dy*dy + dz*dz;
      float mm = fminf(mind[i], d);
      mind[i] = mm;
      if (mm > bv || (mm == bv && i < bi)) { bv = mm; bi = i; }
    }
    const int lane = tid & 31;
    #pragma unroll
    for (int off = 16; off >= 1; off >>= 1) {
      float ov = __shfl_down(bv, off, 32);
      int   oi = __shfl_down(bi, off, 32);
      if (ov > bv || (ov == bv && oi < bi)) { bv = ov; bi = oi; }
    }
    if (lane == 0) { redv[tid >> 5] = bv; redi[tid >> 5] = bi; }
    __syncthreads();
    if (tid == 0) {
      float Bv = redv[0]; int Bi = redi[0];
      for (int w = 1; w < NT / 32; ++w)
        if (redv[w] > Bv || (redv[w] == Bv && redi[w] < Bi)) { Bv = redv[w]; Bi = redi[w]; }
      idx[b * M + it] = Bi;
      float px = P[Bi*3+0], py = P[Bi*3+1], pz = P[Bi*3+2];
      centers[((size_t)b * M + it) * 3 + 0] = px;
      centers[((size_t)b * M + it) * 3 + 1] = py;
      centers[((size_t)b * M + it) * 3 + 2] = pz;
      cpt[0] = px; cpt[1] = py; cpt[2] = pz;
    }
    __syncthreads();
  }
}

// ---------------------------------------------------------------------------
// k-nearest: one wave32 per query; per-lane sorted insertion list, then
// K rounds of wave argmin (shuffle butterfly) to merge.
// ---------------------------------------------------------------------------
template<int K>
__global__ __launch_bounds__(256)
void knn_kernel(const float* __restrict__ q, const float* __restrict__ p,
                int M, int Nn, int total,
                int* __restrict__ nidx, float* __restrict__ nd2)
{
  const int lane = threadIdx.x;
  const int wid  = blockIdx.x * blockDim.y + threadIdx.y;
  if (wid >= total) return;
  const int b = wid / M;
  const float* P = p + (size_t)b * Nn * 3;
  const float qx = q[(size_t)wid*3+0], qy = q[(size_t)wid*3+1], qz = q[(size_t)wid*3+2];
  float lv[K]; int li[K];
  #pragma unroll
  for (int t = 0; t < K; ++t) { lv[t] = 3e38f; li[t] = 0x7FFFFFFF; }
  for (int j = lane; j < Nn; j += 32) {
    float dx = P[j*3+0]-qx, dy = P[j*3+1]-qy, dz = P[j*3+2]-qz;
    float dd = dx*dx + dy*dy + dz*dz;
    int   id = j;
    #pragma unroll
    for (int t = 0; t < K; ++t) {
      if (dd < lv[t] || (dd == lv[t] && id < li[t])) {
        float tv = lv[t]; lv[t] = dd; dd = tv;
        int   ti = li[t]; li[t] = id; id = ti;
      }
    }
  }
  #pragma unroll
  for (int t = 0; t < K; ++t) {
    float bv = lv[0]; int bi = li[0]; int bl = lane;
    #pragma unroll
    for (int off = 16; off >= 1; off >>= 1) {
      float ov = __shfl_xor(bv, off, 32);
      int   oi = __shfl_xor(bi, off, 32);
      int   ol = __shfl_xor(bl, off, 32);
      if (ov < bv || (ov == bv && oi < bi)) { bv = ov; bi = oi; bl = ol; }
    }
    if (lane == 0) { nd2[(size_t)wid*K + t] = bv; nidx[(size_t)wid*K + t] = bi; }
    if (lane == bl) {
      #pragma unroll
      for (int s2 = 0; s2 < K - 1; ++s2) { lv[s2] = lv[s2+1]; li[s2] = li[s2+1]; }
      lv[K-1] = 3e38f; li[K-1] = 0x7FFFFFFF;
    }
  }
}

// ---------------------------------------------------------------------------
// f16 fill kernels, 8 halves (128 bits) per thread. Kpad % 8 == 0 always.
// ---------------------------------------------------------------------------
__global__ void gather_group_v8(const float* __restrict__ feat, int C, int Nn,
                                const float* __restrict__ pos,
                                const float* __restrict__ centers,
                                const int* __restrict__ nidx,
                                int M, int Kpad, long long total8,
                                v8h* __restrict__ out)
{
  for (long long e = (long long)blockIdx.x * blockDim.x + threadIdx.x; e < total8;
       e += (long long)gridDim.x * blockDim.x) {
    long long base = e * 8;
    int col0 = (int)(base % Kpad);
    long long row = base / Kpad;
    int kn = (int)(row & 15);
    long long cm = row >> 4;            // b*M + m
    int b = (int)(cm / M);
    int pidx = nidx[cm * KNB + kn];
    v8h o;
    #pragma unroll
    for (int j = 0; j < 8; ++j) {
      int col = col0 + j;
      float v = 0.f;
      if (col < C) v = feat[((long long)b * Nn + pidx) * C + col];
      else if (col < C + 3) {
        int d = col - C;
        v = pos[((long long)b * Nn + pidx) * 3 + d] - centers[cm * 3 + d];
      }
      o[j] = (_Float16)v;
    }
    out[e] = o;
  }
}

__global__ void interp_concat_v8(const float* __restrict__ src, int Csrc, int Np,
                                 const int* __restrict__ nidx, const float* __restrict__ nd2, int K,
                                 const float* __restrict__ skip, int Cskip,
                                 int M, int Kpad, long long total8,
                                 v8h* __restrict__ out)
{
  for (long long e = (long long)blockIdx.x * blockDim.x + threadIdx.x; e < total8;
       e += (long long)gridDim.x * blockDim.x) {
    long long base = e * 8;
    int col0 = (int)(base % Kpad);
    long long row = base / Kpad;
    int b = (int)(row / M);
    float w[3]; int id[3]; float wsum = 0.f;
    for (int k = 0; k < K; ++k) {
      float d = nd2[row * K + k];
      w[k] = 1.f / fmaxf(d, 1e-16f);
      id[k] = nidx[row * K + k];
      wsum += w[k];
    }
    v8h o;
    #pragma unroll
    for (int j = 0; j < 8; ++j) {
      int col = col0 + j;
      float v = 0.f;
      if (col < Csrc) {
        float acc = 0.f;
        for (int k = 0; k < K; ++k)
          acc += w[k] * src[((long long)b * Np + id[k]) * Csrc + col];
        v = acc / wsum;
      } else if (col < Csrc + Cskip) {
        v = skip[row * Cskip + (col - Csrc)];
      }
      o[j] = (_Float16)v;
    }
    out[e] = o;
  }
}

__global__ void concat2_v8(const float* __restrict__ a, int Ca,
                           const float* __restrict__ bsrc, int Cb,
                           int Kpad, long long total8, v8h* __restrict__ out)
{
  for (long long e = (long long)blockIdx.x * blockDim.x + threadIdx.x; e < total8;
       e += (long long)gridDim.x * blockDim.x) {
    long long base = e * 8;
    int col0 = (int)(base % Kpad);
    long long row = base / Kpad;
    v8h o;
    #pragma unroll
    for (int j = 0; j < 8; ++j) {
      int col = col0 + j;
      float v = 0.f;
      if (col < Ca) v = a[row * Ca + col];
      else if (col < Ca + Cb) v = bsrc[row * Cb + (col - Ca)];
      o[j] = (_Float16)v;
    }
    out[e] = o;
  }
}

__global__ void concat_bcast_v8(const float* __restrict__ g, int Ca,
                                const float* __restrict__ skp, int Cb,
                                int M, int Kpad, long long total8,
                                v8h* __restrict__ out)
{
  for (long long e = (long long)blockIdx.x * blockDim.x + threadIdx.x; e < total8;
       e += (long long)gridDim.x * blockDim.x) {
    long long base = e * 8;
    int col0 = (int)(base % Kpad);
    long long row = base / Kpad;
    int b = (int)(row / M);
    v8h o;
    #pragma unroll
    for (int j = 0; j < 8; ++j) {
      int col = col0 + j;
      float v = 0.f;
      if (col < Ca) v = g[(long long)b * Ca + col];
      else if (col < Ca + Cb) v = skp[row * Cb + (col - Ca)];
      o[j] = (_Float16)v;
    }
    out[e] = o;
  }
}

__global__ void maxpool_rows(const float* __restrict__ X, int M, int C,
                             float* __restrict__ out)
{
  int c = blockIdx.x * blockDim.x + threadIdx.x;
  int b = blockIdx.y;
  if (c >= C) return;
  float mx = -3e38f;
  for (int m = 0; m < M; ++m) mx = fmaxf(mx, X[((size_t)b * M + m) * C + c]);
  out[(size_t)b * C + c] = mx;
}

// ---------------------------------------------------------------------------
// Weight f32 [K,C] -> f16 packed directly in the WMMA B-fragment layout:
//   wp[((ct*KT + kt)*32 + lane)*16 + e]  (e = fragment element index)
// so each lane's fragment is 32 contiguous bytes (2x global_load_b128).
// ---------------------------------------------------------------------------
__global__ void convert_w_pack(const float* __restrict__ W, int K, int C,
                               int KT, long long total, _Float16* __restrict__ out)
{
  for (long long o = (long long)blockIdx.x * blockDim.x + threadIdx.x; o < total;
       o += (long long)gridDim.x * blockDim.x) {
    int e    = (int)(o & 15);
    int lane = (int)((o >> 4) & 31);
    long long rest = o >> 9;
    int kt = (int)(rest % KT);
    int ct = (int)(rest / KT);
    int i = e >> 1, p = e & 1;
    int kk = ((i >> 2) << 4) + ((lane >> 4) << 3) + ((i & 3) << 1) + p;
    int k  = kt * 32 + kk;
    int col = ct * 16 + (lane & 15);
    out[o] = (k < K && col < C) ? (_Float16)W[(long long)k * C + col] : (_Float16)0.f;
  }
}

// ---------------------------------------------------------------------------
// WMMA GEMM, 2x2 tiles (32x32 output) per wave, K stepped by 32.
// A fragments: two unconditional b128 loads per row tile (X row-padded to 32).
// B fragments: 32B contiguous from packed weights.
// Epilogue: +bias, optional ReLU, folded-BN affine; optional fused masked max
// over each 16-row neighbor group via shfl_xor(16).
// ---------------------------------------------------------------------------
__global__ __launch_bounds__(256)
void gemm_wmma(const _Float16* __restrict__ X, const _Float16* __restrict__ W,
               int R, int Kpad, int Cout, int Cp,
               const float* __restrict__ bias, const float* __restrict__ gamma,
               const float* __restrict__ beta,
               int relu, int pool, const float* __restrict__ vd2, float r2,
               float* __restrict__ Yf32, _Float16* __restrict__ Yf16,
               int CT2, int waves)
{
  const int wid = blockIdx.x * blockDim.y + threadIdx.y;
  if (wid >= waves) return;
  const int rt2 = wid / CT2, ct2 = wid % CT2;
  const int lane = threadIdx.x;
  const int m = lane & 15, half = lane >> 4;
  const int KT = Kpad >> 5;

  const _Float16* xr0 = X + (long long)(rt2 * 32 + m) * Kpad + (half << 3);
  const _Float16* xr1 = xr0 + (long long)16 * Kpad;
  const v16h* WP = (const v16h*)W;
  const long long wb0 = ((long long)(ct2 * 2 + 0) * KT) * 32 + lane;
  const long long wb1 = ((long long)(ct2 * 2 + 1) * KT) * 32 + lane;

  v8f acc[2][2] = {};
  for (int kt = 0; kt < KT; ++kt) {
    const int k0 = kt << 5;
    v8h a0lo = *(const v8h*)(xr0 + k0);
    v8h a0hi = *(const v8h*)(xr0 + k0 + 16);
    v8h a1lo = *(const v8h*)(xr1 + k0);
    v8h a1hi = *(const v8h*)(xr1 + k0 + 16);
    v16h a0 = __builtin_shufflevector(a0lo, a0hi, 0,1,2,3,4,5,6,7,8,9,10,11,12,13,14,15);
    v16h a1 = __builtin_shufflevector(a1lo, a1hi, 0,1,2,3,4,5,6,7,8,9,10,11,12,13,14,15);
    v16h b0 = WP[wb0 + (long long)kt * 32];
    v16h b1 = WP[wb1 + (long long)kt * 32];
    acc[0][0] = __builtin_amdgcn_wmma_f32_16x16x32_f16(false, a0, false, b0, (short)0, acc[0][0], false, false);
    acc[0][1] = __builtin_amdgcn_wmma_f32_16x16x32_f16(false, a0, false, b1, (short)0, acc[0][1], false, false);
    acc[1][0] = __builtin_amdgcn_wmma_f32_16x16x32_f16(false, a1, false, b0, (short)0, acc[1][0], false, false);
    acc[1][1] = __builtin_amdgcn_wmma_f32_16x16x32_f16(false, a1, false, b1, (short)0, acc[1][1], false, false);
  }

  #pragma unroll
  for (int tc = 0; tc < 2; ++tc) {
    const int colg = (ct2 * 2 + tc) * 16 + m;
    float s = 1.f, t = 0.f, bb = 0.f;
    if (colg < Cout) {
      if (bias)  bb = bias[colg];
      if (gamma) { s = BN_SCALE_F * gamma[colg]; t = beta[colg]; }
    }
    #pragma unroll
    for (int tr = 0; tr < 2; ++tr) {
      const int rtile = rt2 * 2 + tr;
      float vals[8];
      #pragma unroll
      for (int r = 0; r < 8; ++r) {
        float v = acc[tr][tc][r] + bb;
        if (relu)  v = fmaxf(v, 0.f);
        if (gamma) v = v * s + t;
        vals[r] = v;
      }
      if (!pool) {
        #pragma unroll
        for (int r = 0; r < 8; ++r) {
          int rowg = rtile * 16 + r + (half << 3);   // C/D layout: VGPR r -> M=r / M=r+8
          if (rowg < R && colg < Cout) {
            long long o = (long long)rowg * Cout + colg;
            if (Yf32) Yf32[o] = vals[r];
            if (Yf16) Yf16[o] = (_Float16)vals[r];
          }
        }
      } else if (rtile * 16 < R) {                   // one 16-row neighbor group
        float mx = -1e10f;
        #pragma unroll
        for (int r = 0; r < 8; ++r) {
          int rowg = rtile * 16 + r + (half << 3);
          bool ok = vd2 ? (vd2[rowg] <= r2) : true;
          mx = fmaxf(mx, ok ? vals[r] : -1e10f);
        }
        mx = fmaxf(mx, __shfl_xor(mx, 16, 32));
        if (half == 0 && colg < Cout) {
          long long o = (long long)rtile * Cout + colg;
          if (Yf32) Yf32[o] = mx;
          if (Yf16) Yf16[o] = (_Float16)mx;
        }
      }
    }
  }
}

static void gemm(hipStream_t s, const _Float16* X, const _Float16* W,
                 int R, int Kp, int Cout, int Cp,
                 const void* bias, const void* gamma, const void* beta,
                 int relu, int pool, const float* vd2, float r2,
                 float* Yf32, _Float16* Yf16)
{
  int RT2 = (R + 31) / 32;
  int CT2 = (Cout + 31) / 32;
  int waves = RT2 * CT2;
  gemm_wmma<<<(waves + 7) / 8, dim3(32, 8), 0, s>>>(
      X, W, R, Kp, Cout, Cp,
      (const float*)bias, (const float*)gamma, (const float*)beta,
      relu, pool, vd2, r2, Yf32, Yf16, CT2, waves);
}

// ---------------------------------------------------------------------------
extern "C" void kernel_launch(void* const* d_in, const int* in_sizes, int n_in,
                              void* d_out, int out_size, void* d_ws, size_t ws_size,
                              hipStream_t stream)
{
  const float* x   = (const float*)d_in[0];   // [8,4096,6]
  const float* pos = (const float*)d_in[1];   // [8,4096,3]

  struct WInfo { int di; int K; int C; int Kp; int Cp; _Float16* w; };
  WInfo wt[14] = {
    { 2,    9,   64,   32,   64, nullptr},  // sa1 l0
    { 6,   64,  128,   64,  128, nullptr},  // sa1 l1
    {10,  131,  128,  160,  128, nullptr},  // sa2 l0
    {14,  128,  256,  128,  256, nullptr},  // sa2 l1
    {18,  259,  512,  288,  512, nullptr},  // sa3 l0
    {22,  512, 1024,  512, 1024, nullptr},  // sa3 l1
    {26, 1280,  256, 1280,  256, nullptr},  // fp3 l0
    {30,  256,  256,  256,  256, nullptr},  // fp3 l1
    {34,  384,  256,  384,  256, nullptr},  // fp2 l0
    {38,  256,  128,  256,  128, nullptr},  // fp2 l1
    {42,  134,  128,  160,  128, nullptr},  // fp1 l0
    {46,  128,  128,  128,  128, nullptr},  // fp1 l1
    {50,  128,  128,  128,  128, nullptr},  // lin1
    {52,  128,   13,  128,   32, nullptr},  // lin3 (Cp padded to 32 for 2-wide tiles)
  };

  char* base = (char*)d_ws;
  size_t off = 0;
  auto alloc = [&](size_t bytes) -> void* {
    void* p = base + off;
    off = (off + bytes + 255) & ~(size_t)255;
    return p;
  };
  auto rpad = [](long long r) -> long long { return (r + 31) & ~31LL; };

  for (int i = 0; i < 14; ++i) {
    long long tot = (long long)wt[i].Kp * wt[i].Cp;
    wt[i].w = (_Float16*)alloc((size_t)tot * sizeof(_Float16));
    convert_w_pack<<<(int)((tot + 255) / 256), 256, 0, stream>>>(
        (const float*)d_in[wt[i].di], wt[i].K, wt[i].C, wt[i].Kp >> 5, tot, wt[i].w);
  }

  int*   idx1  = (int*)  alloc(sizeof(int)   * BATCH * M1);
  float* pos1  = (float*)alloc(sizeof(float) * BATCH * M1 * 3);
  int*   idx2  = (int*)  alloc(sizeof(int)   * BATCH * M2);
  float* pos2  = (float*)alloc(sizeof(float) * BATCH * M2 * 3);
  int*   nidx1 = (int*)  alloc(sizeof(int)   * (size_t)BATCH * M1 * KNB);
  float* nd21  = (float*)alloc(sizeof(float) * (size_t)BATCH * M1 * KNB);
  int*   nidx2 = (int*)  alloc(sizeof(int)   * (size_t)BATCH * M2 * KNB);
  float* nd22  = (float*)alloc(sizeof(float) * (size_t)BATCH * M2 * KNB);

  const long long R1g = (long long)BATCH * M1 * KNB;   // 104960 (mult of 32)
  const long long R2g = (long long)BATCH * M2 * KNB;   // 26240  (mult of 32)
  const long long Rc2 = (long long)BATCH * M2;         // 1640 -> pad 1664
  const long long Rc1 = (long long)BATCH * M1;         // 6560  (mult of 32)
  const long long Rn  = (long long)BATCH * NPTS;       // 32768 (mult of 32)

  _Float16* g1x  = (_Float16*)alloc(2 * rpad(R1g) * 32);
  _Float16* h1   = (_Float16*)alloc(2 * rpad(R1g) * 64);
  float*    x1   = (float*)   alloc(4 * Rc1 * 128);
  _Float16* g2x  = (_Float16*)alloc(2 * rpad(R2g) * 160);
  _Float16* h2   = (_Float16*)alloc(2 * rpad(R2g) * 128);
  float*    x2   = (float*)   alloc(4 * Rc2 * 256);
  _Float16* s3in = (_Float16*)alloc(2 * rpad(Rc2) * 288);
  _Float16* h3   = (_Float16*)alloc(2 * rpad(Rc2) * 512);
  float*    h3o  = (float*)   alloc(4 * Rc2 * 1024);
  float*    x3   = (float*)   alloc(4 * BATCH * 1024);
  _Float16* f3in = (_Float16*)alloc(2 * rpad(Rc2) * 1280);
  _Float16* f3h  = (_Float16*)alloc(2 * rpad(Rc2) * 256);
  float*    y2   = (float*)   alloc(4 * Rc2 * 256);
  int*    nidxF2 = (int*)     alloc(sizeof(int)   * Rc1 * 3);
  float*  nd2F2  = (float*)   alloc(sizeof(float) * Rc1 * 3);
  _Float16* f2in = (_Float16*)alloc(2 * rpad(Rc1) * 384);
  _Float16* f2h  = (_Float16*)alloc(2 * rpad(Rc1) * 256);
  float*    y1   = (float*)   alloc(4 * Rc1 * 128);
  int*    nidxF1 = (int*)     alloc(sizeof(int)   * Rn * 3);
  float*  nd2F1  = (float*)   alloc(sizeof(float) * Rn * 3);
  _Float16* f1in = (_Float16*)alloc(2 * rpad(Rn) * 160);
  _Float16* f1h  = (_Float16*)alloc(2 * rpad(Rn) * 128);
  _Float16* y0   = (_Float16*)alloc(2 * rpad(Rn) * 128);
  _Float16* lh   = (_Float16*)alloc(2 * rpad(Rn) * 128);

  // ---------------- SA1 ----------------
  fps_kernel<256><<<BATCH, 256, 0, stream>>>(pos, NPTS, M1, idx1, pos1);
  {
    int total = BATCH * M1;
    knn_kernel<KNB><<<(total + 7) / 8, dim3(32, 8), 0, stream>>>(
        pos1, pos, M1, NPTS, total, nidx1, nd21);
  }
  {
    long long tot8 = R1g * 32 / 8;
    gather_group_v8<<<(int)((tot8 + 255) / 256), 256, 0, stream>>>(
        x, CIN, NPTS, pos, pos1, nidx1, M1, 32, tot8, (v8h*)g1x);
  }
  gemm(stream, g1x, wt[0].w, (int)R1g, 32, 64, 64,
       d_in[3], d_in[4], d_in[5], 1, 0, nullptr, 0.f, nullptr, h1);
  gemm(stream, h1, wt[1].w, (int)R1g, 64, 128, 128,
       d_in[7], d_in[8], d_in[9], 1, 1, nd21, 0.01f, x1, nullptr);

  // ---------------- SA2 ----------------
  fps_kernel<256><<<BATCH, 256, 0, stream>>>(pos1, M1, M2, idx2, pos2);
  {
    int total = BATCH * M2;
    knn_kernel<KNB><<<(total + 7) / 8, dim3(32, 8), 0, stream>>>(
        pos2, pos1, M2, M1, total, nidx2, nd22);
  }
  {
    long long tot8 = R2g * 160 / 8;
    gather_group_v8<<<(int)((tot8 + 255) / 256), 256, 0, stream>>>(
        x1, 128, M1, pos1, pos2, nidx2, M2, 160, tot8, (v8h*)g2x);
  }
  gemm(stream, g2x, wt[2].w, (int)R2g, 160, 128, 128,
       d_in[11], d_in[12], d_in[13], 1, 0, nullptr, 0.f, nullptr, h2);
  gemm(stream, h2, wt[3].w, (int)R2g, 128, 256, 256,
       d_in[15], d_in[16], d_in[17], 1, 1, nd22, 0.04f, x2, nullptr);

  // ---------------- SA3 (global) ----------------
  {
    long long tot8 = Rc2 * 288 / 8;
    concat2_v8<<<(int)((tot8 + 255) / 256), 256, 0, stream>>>(
        x2, 256, pos2, 3, 288, tot8, (v8h*)s3in);
  }
  gemm(stream, s3in, wt[4].w, (int)Rc2, 288, 512, 512,
       d_in[19], d_in[20], d_in[21], 1, 0, nullptr, 0.f, nullptr, h3);
  gemm(stream, h3, wt[5].w, (int)Rc2, 512, 1024, 1024,
       d_in[23], d_in[24], d_in[25], 1, 0, nullptr, 0.f, h3o, nullptr);
  maxpool_rows<<<dim3((1024 + 255) / 256, BATCH), 256, 0, stream>>>(h3o, M2, 1024, x3);

  // ---------------- FP3 (k=1 -> broadcast) ----------------
  {
    long long tot8 = Rc2 * 1280 / 8;
    concat_bcast_v8<<<(int)((tot8 + 255) / 256), 256, 0, stream>>>(
        x3, 1024, x2, 256, M2, 1280, tot8, (v8h*)f3in);
  }
  gemm(stream, f3in, wt[6].w, (int)Rc2, 1280, 256, 256,
       d_in[27], d_in[28], d_in[29], 1, 0, nullptr, 0.f, nullptr, f3h);
  gemm(stream, f3h, wt[7].w, (int)Rc2, 256, 256, 256,
       d_in[31], d_in[32], d_in[33], 1, 0, nullptr, 0.f, y2, nullptr);

  // ---------------- FP2 ----------------
  {
    int total = BATCH * M1;
    knn_kernel<3><<<(total + 7) / 8, dim3(32, 8), 0, stream>>>(
        pos1, pos2, M1, M2, total, nidxF2, nd2F2);
  }
  {
    long long tot8 = Rc1 * 384 / 8;
    interp_concat_v8<<<(int)((tot8 + 255) / 256), 256, 0, stream>>>(
        y2, 256, M2, nidxF2, nd2F2, 3, x1, 128, M1, 384, tot8, (v8h*)f2in);
  }
  gemm(stream, f2in, wt[8].w, (int)Rc1, 384, 256, 256,
       d_in[35], d_in[36], d_in[37], 1, 0, nullptr, 0.f, nullptr, f2h);
  gemm(stream, f2h, wt[9].w, (int)Rc1, 256, 128, 128,
       d_in[39], d_in[40], d_in[41], 1, 0, nullptr, 0.f, y1, nullptr);

  // ---------------- FP1 ----------------
  {
    int total = BATCH * NPTS;
    knn_kernel<3><<<(total + 7) / 8, dim3(32, 8), 0, stream>>>(
        pos, pos1, NPTS, M1, total, nidxF1, nd2F1);
  }
  {
    long long tot8 = Rn * 160 / 8;
    interp_concat_v8<<<(int)((tot8 + 255) / 256), 256, 0, stream>>>(
        y1, 128, M1, nidxF1, nd2F1, 3, x, CIN, NPTS, 160, tot8, (v8h*)f1in);
  }
  gemm(stream, f1in, wt[10].w, (int)Rn, 160, 128, 128,
       d_in[43], d_in[44], d_in[45], 1, 0, nullptr, 0.f, nullptr, f1h);
  gemm(stream, f1h, wt[11].w, (int)Rn, 128, 128, 128,
       d_in[47], d_in[48], d_in[49], 1, 0, nullptr, 0.f, nullptr, y0);

  // ---------------- Head ----------------
  gemm(stream, y0, wt[12].w, (int)Rn, 128, 128, 128,
       d_in[51], nullptr, nullptr, 1, 0, nullptr, 0.f, nullptr, lh);
  gemm(stream, lh, wt[13].w, (int)Rn, 128, NCLS, 32,
       d_in[53], nullptr, nullptr, 0, 0, nullptr, 0.f, (float*)d_out, nullptr);
}
